// Pose_71863392797134
// MI455X (gfx1250) — matrix operand order repaired and checked by reading, not compiled, not run on hardware
//
#include <hip/hip_runtime.h>
#include <hip/hip_bf16.h>

typedef __attribute__((ext_vector_type(16))) _Float16 v16h;
typedef __attribute__((ext_vector_type(8)))  _Float16 v8h;
typedef __attribute__((ext_vector_type(8)))  float    v8f;
typedef __attribute__((ext_vector_type(4)))  float    v4f;

#define NW      8                 // waves per block (wave32)
#define TPB     (NW * 32)
#define WSTRIDE 136               // halves per padded row (128 + 8) -> conflict-free LDS
#define NTN     7                 // N tiles: 112 = 7 * 16

__global__ __launch_bounds__(TPB) void pose_mlp_polar_kernel(
    const float* __restrict__ x,  const float* __restrict__ w1, const float* __restrict__ b1,
    const float* __restrict__ w2, const float* __restrict__ b2,
    const float* __restrict__ w3, const float* __restrict__ b3,
    float* __restrict__ out, int nrows)
{
    __shared__ __align__(16) _Float16 w1s[112 * WSTRIDE];   // B-layout: w1s[j*WSTRIDE + k] = w1[j][k]
    __shared__ __align__(16) _Float16 w2s[112 * WSTRIDE];
    __shared__ __align__(16) _Float16 w3s[16  * WSTRIDE];
    __shared__ float b1s[112], b2s[112], b3s[16];
    __shared__ __align__(16) _Float16 hscr[NW * 16 * WSTRIDE]; // per-wave activation scratch (f16)
    __shared__ __align__(16) float    msbuf[NW * 256];         // per-wave 16x16 f32 output tile

    const int tid = threadIdx.x;

    // ---- stage weights/biases into LDS (zero-padded) ----
    for (int i = tid; i < 112 * WSTRIDE; i += TPB) {
        int j = i / WSTRIDE, k = i - j * WSTRIDE;
        w1s[i] = (_Float16)((j < 100 && k < 128) ? w1[j * 128 + k] : 0.f);
        w2s[i] = (_Float16)((j < 100 && k < 100) ? w2[j * 100 + k] : 0.f);
    }
    for (int i = tid; i < 16 * WSTRIDE; i += TPB) {
        int j = i / WSTRIDE, k = i - j * WSTRIDE;
        w3s[i] = (_Float16)((j < 9 && k < 100) ? w3[j * 100 + k] : 0.f);
    }
    for (int i = tid; i < 112; i += TPB) {
        b1s[i] = (i < 100) ? b1[i] : 0.f;
        b2s[i] = (i < 100) ? b2[i] : 0.f;
    }
    if (tid < 16) b3s[tid] = (tid < 9) ? b3[tid] : 0.f;
    {   // zero activation scratch once (pad columns must stay finite-zero forever)
        unsigned* z = (unsigned*)hscr;
        for (int i = tid; i < NW * 16 * WSTRIDE / 2; i += TPB) z[i] = 0u;
    }
    __syncthreads();

    const int lane = tid & 31;
    const int wv   = tid >> 5;
    const int nlo  = lane & 15;     // N / M-low index
    const int hi   = lane >> 4;     // half-wave selector
    _Float16* hs = hscr  + wv * 16 * WSTRIDE;
    float*    ms = msbuf + wv * 256;

    const int NT     = nrows >> 4;          // 16-row tiles
    const int stride = gridDim.x * NW;

    for (int t = blockIdx.x * NW + wv; t < NT; t += stride) {
        const int r0 = t << 4;
        const float* xrow = x + (size_t)(r0 + nlo) * 128;

        // prefetch next x tile into cache (global_prefetch_b8)
        {
            int tn = t + stride;
            if (tn < NT) {
                const char* pn = (const char*)(x + ((size_t)tn << 4) * 128);
                __builtin_prefetch(pn + lane * 256, 0, 0);
            }
        }

        v8f acc[NTN] = {};
        v8f zed = {};

        // ================= GEMM1: h1 = relu(x @ w1^T + b1), K = 128 =================
        #pragma unroll
        for (int kk = 0; kk < 4; ++kk) {
            const float* xp = xrow + 32 * kk + 8 * hi;
            v4f p0 = *(const v4f*)(xp);
            v4f p1 = *(const v4f*)(xp + 4);
            v4f p2 = *(const v4f*)(xp + 16);
            v4f p3 = *(const v4f*)(xp + 20);
            v8f f0 = __builtin_shufflevector(p0, p1, 0,1,2,3,4,5,6,7);
            v8f f1 = __builtin_shufflevector(p2, p3, 0,1,2,3,4,5,6,7);
            v8h h0 = __builtin_convertvector(f0, v8h);
            v8h h1 = __builtin_convertvector(f1, v8h);
            v16h a = __builtin_shufflevector(h0, h1, 0,1,2,3,4,5,6,7,8,9,10,11,12,13,14,15);
            #pragma unroll
            for (int n = 0; n < NTN; ++n) {
                const _Float16* wp = w1s + (16 * n + nlo) * WSTRIDE + 32 * kk + 8 * hi;
                v8h bl0 = *(const v8h*)(wp);
                v8h bl1 = *(const v8h*)(wp + 16);
                v16h bm = __builtin_shufflevector(bl0, bl1, 0,1,2,3,4,5,6,7,8,9,10,11,12,13,14,15);
                acc[n] = __builtin_amdgcn_wmma_f32_16x16x32_f16(false, a, false, bm,
                                                                (short)0, acc[n], false, false);
            }
        }
        // bias + relu, write h1 to LDS in row-major f16 (A-layout source)
        #pragma unroll
        for (int n = 0; n < NTN; ++n) {
            int N = 16 * n + nlo;
            if (N < 100) {
                float bv = b1s[N];
                #pragma unroll
                for (int g = 0; g < 8; ++g) {
                    float v = acc[n][g] + bv;
                    hs[(g + 8 * hi) * WSTRIDE + N] = (_Float16)(v > 0.f ? v : 0.f);
                }
            }
            acc[n] = zed;
        }

        // ================= GEMM2: h2 = tanh(h1 @ w2^T + b2), K = 100 (pad 128) =======
        #pragma unroll
        for (int kk = 0; kk < 4; ++kk) {
            const _Float16* hp = hs + nlo * WSTRIDE + 32 * kk + 8 * hi;
            v8h q0 = *(const v8h*)(hp);
            v8h q1 = *(const v8h*)(hp + 16);
            v16h a = __builtin_shufflevector(q0, q1, 0,1,2,3,4,5,6,7,8,9,10,11,12,13,14,15);
            #pragma unroll
            for (int n = 0; n < NTN; ++n) {
                const _Float16* wp = w2s + (16 * n + nlo) * WSTRIDE + 32 * kk + 8 * hi;
                v8h bl0 = *(const v8h*)(wp);
                v8h bl1 = *(const v8h*)(wp + 16);
                v16h bm = __builtin_shufflevector(bl0, bl1, 0,1,2,3,4,5,6,7,8,9,10,11,12,13,14,15);
                acc[n] = __builtin_amdgcn_wmma_f32_16x16x32_f16(false, a, false, bm,
                                                                (short)0, acc[n], false, false);
            }
        }
        #pragma unroll
        for (int n = 0; n < NTN; ++n) {
            int N = 16 * n + nlo;
            if (N < 100) {
                float bv = b2s[N];
                #pragma unroll
                for (int g = 0; g < 8; ++g)
                    hs[(g + 8 * hi) * WSTRIDE + N] = (_Float16)tanhf(acc[n][g] + bv);
            }
        }

        // ================= GEMM3: m = h2 @ w3^T + b3  (9 cols, one N tile) ===========
        v8f m3 = {};
        #pragma unroll
        for (int kk = 0; kk < 4; ++kk) {
            const _Float16* hp = hs + nlo * WSTRIDE + 32 * kk + 8 * hi;
            v8h q0 = *(const v8h*)(hp);
            v8h q1 = *(const v8h*)(hp + 16);
            v16h a = __builtin_shufflevector(q0, q1, 0,1,2,3,4,5,6,7,8,9,10,11,12,13,14,15);
            const _Float16* wp = w3s + nlo * WSTRIDE + 32 * kk + 8 * hi;
            v8h bl0 = *(const v8h*)(wp);
            v8h bl1 = *(const v8h*)(wp + 16);
            v16h bm = __builtin_shufflevector(bl0, bl1, 0,1,2,3,4,5,6,7,8,9,10,11,12,13,14,15);
            m3 = __builtin_amdgcn_wmma_f32_16x16x32_f16(false, a, false, bm,
                                                        (short)0, m3, false, false);
        }
        // transpose m tile via LDS so one lane owns one row
        {
            float b3v = b3s[nlo];
            #pragma unroll
            for (int g = 0; g < 8; ++g)
                ms[(g + 8 * hi) * 16 + nlo] = m3[g] + b3v;
        }

        // ===== nearest rotation: polar factor via scaled Newton (== U @ Vh of SVD) ===
        if (lane < 16) {
            float X[9];
            #pragma unroll
            for (int j = 0; j < 9; ++j) X[j] = ms[lane * 16 + j];
            #pragma unroll 1
            for (int it = 0; it < 8; ++it) {
                float c00 = X[4]*X[8] - X[5]*X[7];
                float c01 = X[5]*X[6] - X[3]*X[8];
                float c02 = X[3]*X[7] - X[4]*X[6];
                float c10 = X[2]*X[7] - X[1]*X[8];
                float c11 = X[0]*X[8] - X[2]*X[6];
                float c12 = X[1]*X[6] - X[0]*X[7];
                float c20 = X[1]*X[5] - X[2]*X[4];
                float c21 = X[2]*X[3] - X[0]*X[5];
                float c22 = X[0]*X[4] - X[1]*X[3];
                float det = X[0]*c00 + X[1]*c01 + X[2]*c02;
                float sgn = (det >= 0.f) ? 1.f : -1.f;
                float inv = sgn / fmaxf(fabsf(det), 1e-20f);
                float Y[9] = { c00*inv, c01*inv, c02*inv,
                               c10*inv, c11*inv, c12*inv,
                               c20*inv, c21*inv, c22*inv };   // X^{-T}
                float nx = 0.f, ny = 0.f;
                #pragma unroll
                for (int j = 0; j < 9; ++j) { nx += X[j]*X[j]; ny += Y[j]*Y[j]; }
                float mu = sqrtf(sqrtf(ny / fmaxf(nx, 1e-30f)));  // Higham scaling
                mu = fminf(fmaxf(mu, 1e-4f), 1e4f);
                float wa = 0.5f * mu, wb = 0.5f / mu;
                #pragma unroll
                for (int j = 0; j < 9; ++j) X[j] = wa * X[j] + wb * Y[j];
            }
            float* op = out + (size_t)(r0 + lane) * 9;
            #pragma unroll
            for (int j = 0; j < 9; ++j) op[j] = X[j];
        }
    }
}

extern "C" void kernel_launch(void* const* d_in, const int* in_sizes, int n_in,
                              void* d_out, int out_size, void* d_ws, size_t ws_size,
                              hipStream_t stream) {
    (void)n_in; (void)out_size; (void)d_ws; (void)ws_size;
    const float* x  = (const float*)d_in[0];
    const float* w1 = (const float*)d_in[1];
    const float* b1 = (const float*)d_in[2];
    const float* w2 = (const float*)d_in[3];
    const float* b2 = (const float*)d_in[4];
    const float* w3 = (const float*)d_in[5];
    const float* b3 = (const float*)d_in[6];
    float* out = (float*)d_out;

    int nrows = in_sizes[0] / 128;          // 262144
    int nt    = nrows >> 4;                 // 16-row tiles
    int grid  = 512;                        // persistent-ish: amortize weight staging
    int maxb  = (nt + NW - 1) / NW;
    if (grid > maxb) grid = maxb;

    pose_mlp_polar_kernel<<<grid, TPB, 0, stream>>>(x, w1, b1, w2, b2, w3, b3, out, nrows);
}
